// RoSALinear_66700842106980
// MI455X (gfx1250) — compile-verified, matrix operand order checked by reading, not run on hardware
//
#include <hip/hip_runtime.h>

// ---------------------------------------------------------------------------
// RoSA linear for MI455X (gfx1250, wave32):
//   out = X @ (W + delta*mask)^T + b + 2.0 * (X @ A^T) @ B^T
// bf16 hi/lo split GEMM (3x v_wmma_f32_16x16x32_bf16 per tile) with
// double-buffered async global->LDS staging (ASYNCcnt) and conflict-free
// ds_load_b128 fragment reads.
// ---------------------------------------------------------------------------

typedef __attribute__((ext_vector_type(16))) __bf16 v16bf;
typedef __attribute__((ext_vector_type(8)))  __bf16 v8bf;
typedef __attribute__((ext_vector_type(8)))  float  v8f;

#define MM 8192      // B*S
#define NN 4096      // D_OUT
#define KK 4096      // D_IN
#define LRANK 16
#define LSCALE 2.0f  // 32 / 16

// GEMM staging geometry
#define BK      64                   // K elements per stage
#define STRIDE  72                   // LDS row stride in bf16 (64 + 8 pad)
#define ARRELEM (128 * STRIDE)       // one 128xBK tile in LDS (elements)
#define BUFELEM (4 * ARRELEM)        // Xh|Xl|Wh|Wl tiles per stage
#define ARRB    (ARRELEM * 2)        // bytes
#define BUFB    (BUFELEM * 2)        // bytes
#define SMEM_BYTES (2 * BUFB)        // double buffered (147456 B)
#define NSTAGES (KK / BK)            // 64

// ------------------------- prep: W_eff = W + delta*mask, split to bf16 ------
__global__ void prep_w_kernel(const float* __restrict__ W,
                              const float* __restrict__ delta,
                              const unsigned char* __restrict__ mask,
                              __bf16* __restrict__ Wh,
                              __bf16* __restrict__ Wl) {
  size_t i = (size_t)blockIdx.x * blockDim.x + threadIdx.x;
  float f = W[i];
  if (mask[i]) f += delta[i];
  __bf16 h = (__bf16)f;
  Wh[i] = h;
  Wl[i] = (__bf16)(f - (float)h);
}

// ------------------------- prep: X split to bf16 ----------------------------
__global__ void prep_x_kernel(const float* __restrict__ x,
                              __bf16* __restrict__ Xh,
                              __bf16* __restrict__ Xl) {
  size_t i = (size_t)blockIdx.x * blockDim.x + threadIdx.x;
  float f = x[i];
  __bf16 h = (__bf16)f;
  Xh[i] = h;
  Xl[i] = (__bf16)(f - (float)h);
}

// ------------------------- LoRA: t = X @ A^T  (M x 16, fp32) ----------------
__global__ void lora_t_kernel(const float* __restrict__ x,
                              const float* __restrict__ A,
                              float* __restrict__ t) {
  const int m     = blockIdx.x;
  const int r     = threadIdx.x & 15;
  const int slice = threadIdx.x >> 4;
  const float* xr = x + (size_t)m * KK + slice * 256;
  const float* ar = A + (size_t)r * KK + slice * 256;
  float s = 0.f;
#pragma unroll 8
  for (int k = 0; k < 256; ++k) s += xr[k] * ar[k];
  __shared__ float red[256];
  red[threadIdx.x] = s;
  __syncthreads();
  if (threadIdx.x < LRANK) {
    float tot = 0.f;
#pragma unroll
    for (int i = 0; i < 16; ++i) tot += red[i * 16 + threadIdx.x];
    t[(size_t)m * LRANK + threadIdx.x] = tot;
  }
}

// ------------------------- async global -> LDS copy, 16B per lane -----------
static __device__ __forceinline__ void async_cp16(unsigned lds_byte_off,
                                                  const __bf16* g) {
  // VDST = per-lane LDS byte offset, VADDR = per-lane 64-bit global address.
  asm volatile("global_load_async_to_lds_b128 %0, %1, off"
               :: "v"(lds_byte_off), "v"((unsigned long long)(uintptr_t)g)
               : "memory");
}

#define WAIT_ASYNC(n) asm volatile("s_wait_asynccnt %0" :: "n"(n) : "memory")

// ------------------------- LDS fragment loader ------------------------------
// 16-bit A/B fragment (lane L: hf=L>>4, row=L&15): elems 0..7 = K+hf*8+{0..7},
// elems 8..15 = K+16+hf*8+{0..7}.  Two 16B ds_load_b128 chunks.
static __device__ __forceinline__ v16bf load_frag(const __bf16* p) {
  v8bf c0 = *reinterpret_cast<const v8bf*>(p);
  v8bf c1 = *reinterpret_cast<const v8bf*>(p + 16);
  v16bf r;
#pragma unroll
  for (int i = 0; i < 8; ++i) { r[i] = c0[i]; r[i + 8] = c1[i]; }
  return r;
}

// ------------------------- main GEMM + fused epilogue -----------------------
// Block tile 128x128, 256 threads = 8 waves in 4(M) x 2(N); each wave owns
// 32x64 = 2x4 WMMA tiles.  K pipelined in BK=64 stages through LDS.
__global__ __launch_bounds__(256) void rosa_gemm_kernel(
    const __bf16* __restrict__ Xh, const __bf16* __restrict__ Xl,
    const __bf16* __restrict__ Wh, const __bf16* __restrict__ Wl,
    const float*  __restrict__ bias,
    const float*  __restrict__ loraB,   // [NN, LRANK] row-major
    const float*  __restrict__ t,       // [MM, LRANK] row-major
    float* __restrict__ out) {
  extern __shared__ __bf16 smem[];

  const int tid    = threadIdx.x;
  const int lane   = tid & 31;
  const int wave   = tid >> 5;
  const int row_in = lane & 15;
  const int hf     = lane >> 4;

  const int blockM = blockIdx.y * 128;
  const int blockN = blockIdx.x * 128;
  const int wmloc  = (wave >> 1) * 32;   // wave M origin within block
  const int wnloc  = (wave & 1) * 64;    // wave N origin within block

  // Per-thread copy pattern: 1024 16B-chunks per 128xBK tile, 4 per thread
  // per array; chunk c -> row c>>3, 16B-column c&7.
  const __bf16* src[4] = { Xh + (size_t)blockM * KK,
                           Xl + (size_t)blockM * KK,
                           Wh + (size_t)blockN * KK,
                           Wl + (size_t)blockN * KK };

  // ---- stage issue: 16 async b128 per wave ----
  auto stage_issue = [&](int kk, int buf) {
#pragma unroll
    for (int arr = 0; arr < 4; ++arr) {
#pragma unroll
      for (int i = 0; i < 4; ++i) {
        const int chunk = tid + 256 * i;     // 0..1023
        const int row   = chunk >> 3;
        const int c     = chunk & 7;
        const unsigned lds = (unsigned)(buf * BUFB + arr * ARRB +
                                        row * (STRIDE * 2) + c * 16);
        async_cp16(lds, src[arr] + (size_t)row * KK + kk + c * 8);
      }
    }
  };

  v8f acc[2][4];
#pragma unroll
  for (int mt = 0; mt < 2; ++mt)
#pragma unroll
    for (int nt = 0; nt < 4; ++nt)
#pragma unroll
      for (int j = 0; j < 8; ++j) acc[mt][nt][j] = 0.f;

  stage_issue(0, 0);

#pragma unroll 1
  for (int s = 0; s < NSTAGES; ++s) {
    if (s + 1 < NSTAGES) {
      stage_issue((s + 1) * BK, (s + 1) & 1);
      WAIT_ASYNC(16);               // stage s's 16 ops retired (in-order)
    } else {
      WAIT_ASYNC(0);
    }
    __syncthreads();

    const __bf16* sb  = smem + (s & 1) * BUFELEM;
    const __bf16* sXh = sb;
    const __bf16* sXl = sb + ARRELEM;
    const __bf16* sWh = sb + 2 * ARRELEM;
    const __bf16* sWl = sb + 3 * ARRELEM;

#pragma unroll
    for (int ks = 0; ks < 2; ++ks) {        // two 32-K WMMA steps per stage
      const int kf = ks * 32 + hf * 8;

      v16bf ah[2], al[2];
#pragma unroll
      for (int mt = 0; mt < 2; ++mt) {
        const int off = (wmloc + mt * 16 + row_in) * STRIDE + kf;
        ah[mt] = load_frag(sXh + off);
        al[mt] = load_frag(sXl + off);
      }
      v16bf bh[4], bl[4];
#pragma unroll
      for (int nt = 0; nt < 4; ++nt) {
        const int off = (wnloc + nt * 16 + row_in) * STRIDE + kf;
        bh[nt] = load_frag(sWh + off);
        bl[nt] = load_frag(sWl + off);
      }

#pragma unroll
      for (int mt = 0; mt < 2; ++mt) {
#pragma unroll
        for (int nt = 0; nt < 4; ++nt) {
          acc[mt][nt] = __builtin_amdgcn_wmma_f32_16x16x32_bf16(
              false, ah[mt], false, bh[nt], (short)0, acc[mt][nt], false, false);
          acc[mt][nt] = __builtin_amdgcn_wmma_f32_16x16x32_bf16(
              false, ah[mt], false, bl[nt], (short)0, acc[mt][nt], false, false);
          acc[mt][nt] = __builtin_amdgcn_wmma_f32_16x16x32_bf16(
              false, al[mt], false, bh[nt], (short)0, acc[mt][nt], false, false);
        }
      }
    }
    __syncthreads();   // protect buffer (s&1) before it is re-filled
  }

  // Epilogue: out = acc + bias[n] + LSCALE * dot(t[m,:], loraB[n,:]).
  // C/D layout: element j of v8f -> M = hf*8 + j, N = row_in (within tile).
#pragma unroll
  for (int nt = 0; nt < 4; ++nt) {
    const int n = blockN + wnloc + nt * 16 + row_in;
    float lb[LRANK];
#pragma unroll
    for (int r = 0; r < LRANK; ++r) lb[r] = loraB[(size_t)n * LRANK + r];
    const float bn = bias[n];
#pragma unroll
    for (int mt = 0; mt < 2; ++mt) {
#pragma unroll
      for (int j = 0; j < 8; ++j) {
        const int m = blockM + wmloc + mt * 16 + hf * 8 + j;
        float lora = 0.f;
#pragma unroll
        for (int r = 0; r < LRANK; ++r) lora += t[(size_t)m * LRANK + r] * lb[r];
        out[(size_t)m * NN + n] = acc[mt][nt][j] + bn + LSCALE * lora;
      }
    }
  }
}

// ---------------------------------------------------------------------------
extern "C" void kernel_launch(void* const* d_in, const int* in_sizes, int n_in,
                              void* d_out, int out_size, void* d_ws, size_t ws_size,
                              hipStream_t stream) {
  const float* x      = (const float*)d_in[0];          // [8192, 4096]
  const float* W      = (const float*)d_in[1];          // [4096, 4096]
  const float* b      = (const float*)d_in[2];          // [4096]
  const float* lora_A = (const float*)d_in[3];          // [16, 4096]
  const float* lora_B = (const float*)d_in[4];          // [4096, 16]
  const float* delta  = (const float*)d_in[5];          // [4096, 4096]
  const unsigned char* mask = (const unsigned char*)d_in[6]; // bool [4096,4096]
  float* out = (float*)d_out;

  // Workspace layout: Xh 64MiB | Xl 64MiB | Wh 32MiB | Wl 32MiB | t 512KiB
  __bf16* Xh = (__bf16*)d_ws;
  __bf16* Xl = Xh + (size_t)MM * KK;
  __bf16* Wh = Xl + (size_t)MM * KK;
  __bf16* Wl = Wh + (size_t)NN * KK;
  float*  t  = (float*)(Wl + (size_t)NN * KK);

  prep_w_kernel<<<(NN * (size_t)KK) / 256, 256, 0, stream>>>(W, delta, mask, Wh, Wl);
  prep_x_kernel<<<(MM * (size_t)KK) / 256, 256, 0, stream>>>(x, Xh, Xl);
  lora_t_kernel<<<MM, 256, 0, stream>>>(x, lora_A, t);

  dim3 grid(NN / 128, MM / 128);
  rosa_gemm_kernel<<<grid, 256, SMEM_BYTES, stream>>>(Xh, Xl, Wh, Wl, b, lora_B, t, out);
}
